// QFire_48060684042990
// MI455X (gfx1250) — 8-bit matrix operand order repaired and checked by reading, compiled, not run on hardware
//
#include <hip/hip_runtime.h>
#include <hip/hip_bf16.h>
#include <stdint.h>

typedef int v8i __attribute__((ext_vector_type(8)));
typedef int v4i __attribute__((ext_vector_type(4)));

#define NB    32
#define CIN   128
#define HDIM  56
#define HW    3136
#define SQ    32
#define EXP   128
#define NCOL  (NB * HW)
#define NTILE (NCOL / 16)

#define OFF_XQ   0
#define SZ_XQ    (NCOL * CIN)
#define OFF_HQ   (OFF_XQ + SZ_XQ)
#define SZ_HQ    (NCOL * SQ)
#define OFF_WSQ  (OFF_HQ + SZ_HQ)
#define SZ_WSQ   (2 * 2 * 1024)
#define OFF_WE1  (OFF_WSQ + SZ_WSQ)
#define SZ_WE1   (8 * 1024)
#define OFF_WE3  (OFF_WE1 + SZ_WE1)
#define SZ_WE3   (9 * 8 * 1024)
#define OFF_BSQ  (OFF_WE3 + SZ_WE3)
#define OFF_BE1  (OFF_BSQ + 128)
#define OFF_BE3  (OFF_BE1 + 512)

#define USE_ASYNC_WINDOW 1

#define LDS_OFF(p) ((uint32_t)(uintptr_t)(__attribute__((address_space(3))) const char*)(p))

__device__ __forceinline__ char qw8(float w) {
    float wc = fminf(fmaxf(w, -1.f), 1.f);
    int k = __float2int_rn(wc * 128.f);
    if (k > 127) k = 127;
    return (char)k;
}
__device__ __forceinline__ float qb(float b) {
    float bc = fminf(fmaxf(b, -1.f), 1.f);
    int k = __float2int_rn(bc * 128.f);
    if (k > 127) k = 127;
    return (float)k * (1.f / 128.f);
}

__device__ __forceinline__ v8i make_v8(v4i lo, v4i hi) {
    v8i r;
    r[0] = lo[0]; r[1] = lo[1]; r[2] = lo[2]; r[3] = lo[3];
    r[4] = hi[0]; r[5] = hi[1]; r[6] = hi[2]; r[7] = hi[3];
    return r;
}
__device__ __forceinline__ v8i wmma_iu8(v8i a, v8i b, v8i c) {
    return __builtin_amdgcn_wmma_i32_16x16x64_iu8(true, a, true, b, c, false, false);
}

__global__ __launch_bounds__(256) void pack_weights_kernel(
    const float* __restrict__ w_sq, const float* __restrict__ b_sq,
    const float* __restrict__ w_e1, const float* __restrict__ b_e1,
    const float* __restrict__ w_e3, const float* __restrict__ b_e3,
    char* __restrict__ ws) {
    char* wsq = ws + OFF_WSQ;
    char* we1 = ws + OFF_WE1;
    char* we3 = ws + OFF_WE3;
    float* bsq = (float*)(ws + OFF_BSQ);
    float* be1 = (float*)(ws + OFF_BE1);
    float* be3 = (float*)(ws + OFF_BE3);

    for (int idx = threadIdx.x; idx < SZ_WSQ; idx += blockDim.x) {
        int blk = idx >> 10, r = idx & 1023;
        int mt = blk >> 1, kk = blk & 1;
        int lane = r >> 5, v = (r >> 2) & 7, b = r & 3;
        int M = mt * 16 + (lane & 15);
        int K = kk * 64 + ((v >> 2) << 5) + ((v & 3) << 2) + ((lane >> 4) << 4) + b;
        wsq[idx] = qw8(w_sq[M * CIN + K]);
    }
    for (int idx = threadIdx.x; idx < SZ_WE1; idx += blockDim.x) {
        int mt = idx >> 10, r = idx & 1023;
        int lane = r >> 5, v = (r >> 2) & 7, b = r & 3;
        int M = mt * 16 + (lane & 15);
        int K = ((v >> 2) << 5) + ((v & 3) << 2) + ((lane >> 4) << 4) + b;
        we1[idx] = (K < SQ) ? qw8(w_e1[M * SQ + K]) : (char)0;
    }
    for (int idx = threadIdx.x; idx < SZ_WE3; idx += blockDim.x) {
        int blk = idx >> 10, r = idx & 1023;
        int tap = blk >> 3, mt = blk & 7;
        int lane = r >> 5, v = (r >> 2) & 7, b = r & 3;
        int M = mt * 16 + (lane & 15);
        int K = ((v >> 2) << 5) + ((v & 3) << 2) + ((lane >> 4) << 4) + b;
        we3[idx] = (K < SQ) ? qw8(w_e3[(M * SQ + K) * 9 + tap]) : (char)0;
    }
    for (int i = threadIdx.x; i < SQ;  i += blockDim.x) bsq[i] = qb(b_sq[i]);
    for (int i = threadIdx.x; i < EXP; i += blockDim.x) { be1[i] = qb(b_e1[i]); be3[i] = qb(b_e3[i]); }
}

__global__ __launch_bounds__(256) void quant_x_kernel(const float* __restrict__ x,
                                                      char* __restrict__ xq) {
    __shared__ alignas(16) char sT[64 * 144];
    const int n   = blockIdx.x / (HW / 64);
    const int pb  = blockIdx.x % (HW / 64);
    const int hw0 = pb * 64;
    const int t   = threadIdx.x;

#pragma unroll
    for (int i = 0; i < 32; ++i) {
        int idx = i * 256 + t;
        int c   = idx >> 6;
        int hwL = idx & 63;
        float v = __builtin_nontemporal_load(&x[((size_t)(n * CIN + c)) * HW + hw0 + hwL]);
        float vc = fminf(fmaxf(v, -4.f), 4.f);
        int k = __float2int_rn(vc * 32.f);
        if (k > 127) k = 127;
        sT[hwL * 144 + c] = (char)k;
    }
    __syncthreads();
#pragma unroll
    for (int i = 0; i < 2; ++i) {
        int j   = i * 256 + t;
        int pos = j >> 3, seg = j & 7;
        v4i val = *(const v4i*)&sT[pos * 144 + seg * 16];
        *(v4i*)&xq[((size_t)(n * HW + hw0 + pos)) * CIN + seg * 16] = val;
    }
}

__global__ __launch_bounds__(256) void squeeze_gemm_kernel(char* __restrict__ ws) {
    const char*  xq  = ws + OFF_XQ;
    const char*  wA  = ws + OFF_WSQ;
    const float* bsq = (const float*)(ws + OFF_BSQ);
    char* hq = ws + OFF_HQ;

    __shared__ alignas(16) char sH[8 * 512];
    const int wid  = threadIdx.x >> 5;
    const int lane = threadIdx.x & 31;
    const int tile = blockIdx.x * 8 + wid;
    const int col  = (tile << 4) + (lane & 15);
    const int grp  = lane >> 4;

    v8i A[2][2];
#pragma unroll
    for (int mt = 0; mt < 2; ++mt)
#pragma unroll
        for (int kk = 0; kk < 2; ++kk) {
            const v4i* p = (const v4i*)(wA + (((mt << 1) + kk) << 10) + (lane << 5));
            A[mt][kk] = make_v8(p[0], p[1]);
        }
    const char* xb = xq + (size_t)col * CIN;
    v8i B[2];
#pragma unroll
    for (int kk = 0; kk < 2; ++kk) {
        v4i lo = *(const v4i*)(xb + kk * 64 + grp * 16);
        v4i hi = *(const v4i*)(xb + kk * 64 + 32 + grp * 16);
        B[kk] = make_v8(lo, hi);
    }

    v8i acc[2];
    acc[0] = {}; acc[1] = {};
#pragma unroll
    for (int mt = 0; mt < 2; ++mt)
#pragma unroll
        for (int kk = 0; kk < 2; ++kk)
            acc[mt] = wmma_iu8(A[mt][kk], B[kk], acc[mt]);

    char* my = sH + (wid << 9);
#pragma unroll
    for (int mt = 0; mt < 2; ++mt)
#pragma unroll
        for (int r = 0; r < 8; ++r) {
            int M = mt * 16 + r + (grp << 3);
            float y = (float)acc[mt][r] * (1.f / 4096.f) + bsq[M];
            y = fminf(fmaxf(y, -8.f), 8.f);
            int k = __float2int_rn(y * 16.f);
            if (k > 127) k = 127;
            int h = (k <= 0) ? 0 : ((k >= 16) ? 32 : (k << 1));
            my[(lane & 15) * 32 + M] = (char)h;
        }
    __syncthreads();
    v4i val = *(const v4i*)(my + lane * 16);
    *(v4i*)&hq[((size_t)((tile << 4) + (lane >> 1))) * 32 + (lane & 1) * 16] = val;
}

__device__ __forceinline__ float expand_out(int acc, float bias) {
    float y = (float)acc * (1.f / 4096.f) + bias;
    y = fminf(fmaxf(y, -8.f), 8.f);
    int k = __float2int_rn(y * 16.f);
    if (k > 127) k = 127;
    if (k < 0)   k = 0;
    return (float)k * (1.f / 16.f);
}

__global__ __launch_bounds__(256) void expand_gemm_kernel(char* __restrict__ ws,
                                                          float* __restrict__ out) {
    const char*  hq  = ws + OFF_HQ;
    const char*  wA1 = ws + OFF_WE1;
    const char*  wA3 = ws + OFF_WE3;
    const float* be1 = (const float*)(ws + OFF_BE1);
    const float* be3 = (const float*)(ws + OFF_BE3);

    __shared__ alignas(16) char sB[260 * 16];

    const int tile = blockIdx.x;
    const int n    = (tile << 4) / HW;
    const int hwt  = (tile << 4) % HW;

    for (int c = threadIdx.x; c < 260; c += 256) {
        int i = c >> 1, half = c & 1;
        int hw = hwt - 57 + i;
#if USE_ASYNC_WINDOW
        if ((unsigned)hw < HW) {
            uint32_t lds = LDS_OFF(&sB[c << 4]);
            uint64_t g   = (uint64_t)(uintptr_t)(hq + ((size_t)(n * HW + hw) << 5) + (half << 4));
            asm volatile("global_load_async_to_lds_b128 %0, %1, off"
                         :: "v"(lds), "v"(g) : "memory");
        } else {
            v4i z; z[0] = 0; z[1] = 0; z[2] = 0; z[3] = 0;
            *(v4i*)&sB[c << 4] = z;
        }
#else
        v4i val; val[0] = 0; val[1] = 0; val[2] = 0; val[3] = 0;
        if ((unsigned)hw < HW)
            val = *(const v4i*)(hq + ((size_t)(n * HW + hw) << 5) + (half << 4));
        *(v4i*)&sB[c << 4] = val;
#endif
    }
#if USE_ASYNC_WINDOW
    asm volatile("s_wait_asynccnt 0" ::: "memory");
#endif
    __syncthreads();

    const int mt   = threadIdx.x >> 5;
    const int lane = threadIdx.x & 31;
    const int colL = lane & 15;
    const int grp  = lane >> 4;
    const int hw   = hwt + colL;
    const int h    = hw / HDIM;
    const int w    = hw % HDIM;

    v4i bz; bz[0] = 0; bz[1] = 0; bz[2] = 0; bz[3] = 0;

    v8i acc3; acc3 = {};
#pragma unroll
    for (int tap = 0; tap < 9; ++tap) {
        const int dy = tap / 3 - 1;
        const int dx = tap % 3 - 1;
        v4i blo = bz;
        if ((unsigned)(h + dy) < HDIM && (unsigned)(w + dx) < HDIM) {
            int i = colL + 57 + dy * HDIM + dx;
            blo = *(const v4i*)&sB[i * 32 + grp * 16];
        }
        v8i Bm = make_v8(blo, bz);
        const v4i* pa = (const v4i*)(wA3 + (((tap << 3) + mt) << 10) + (lane << 5));
        v8i Am = make_v8(pa[0], pa[1]);
        acc3 = wmma_iu8(Am, Bm, acc3);
    }

    v8i acc1; acc1 = {};
    {
        v4i blo = *(const v4i*)&sB[(colL + 57) * 32 + grp * 16];
        v8i Bm = make_v8(blo, bz);
        const v4i* pa = (const v4i*)(wA1 + (mt << 10) + (lane << 5));
        v8i Am = make_v8(pa[0], pa[1]);
        acc1 = wmma_iu8(Am, Bm, acc1);
    }

#pragma unroll
    for (int r = 0; r < 8; ++r) {
        int ch = mt * 16 + r + (grp << 3);
        __builtin_nontemporal_store(expand_out(acc1[r], be1[ch]),
                                    &out[((size_t)n * 256 + ch) * HW + hw]);
        __builtin_nontemporal_store(expand_out(acc3[r], be3[ch]),
                                    &out[((size_t)n * 256 + 128 + ch) * HW + hw]);
    }
}

extern "C" void kernel_launch(void* const* d_in, const int* in_sizes, int n_in,
                              void* d_out, int out_size, void* d_ws, size_t ws_size,
                              hipStream_t stream) {
    (void)in_sizes; (void)n_in; (void)out_size; (void)ws_size;
    const float* x    = (const float*)d_in[0];
    const float* w_sq = (const float*)d_in[1];
    const float* b_sq = (const float*)d_in[2];
    const float* w_e1 = (const float*)d_in[3];
    const float* b_e1 = (const float*)d_in[4];
    const float* w_e3 = (const float*)d_in[5];
    const float* b_e3 = (const float*)d_in[6];
    char*  ws  = (char*)d_ws;
    float* out = (float*)d_out;

    pack_weights_kernel<<<1, 256, 0, stream>>>(w_sq, b_sq, w_e1, b_e1, w_e3, b_e3, ws);
    quant_x_kernel<<<NB * (HW / 64), 256, 0, stream>>>(x, ws + OFF_XQ);
    squeeze_gemm_kernel<<<NTILE / 8, 256, 0, stream>>>(ws);
    expand_gemm_kernel<<<NTILE, 256, 0, stream>>>(ws, out);
}
